// AttentionLayer_11476152614983
// MI455X (gfx1250) — matrix operand order repaired and checked by reading, not compiled, not run on hardware
//
#include <hip/hip_runtime.h>
#include <hip/hip_bf16.h>

#define BQ   4
#define DIN  512
#define NTOK 2048
#define DOUT 512

typedef __attribute__((ext_vector_type(16))) __bf16 v16bf;
typedef __attribute__((ext_vector_type(8)))  float  v8f;

union ABFrag {
    v16bf        v;
    unsigned int u[8];
};

// fp32 -> bf16 with round-to-nearest-even
__device__ __forceinline__ unsigned short f2bf(float f) {
    unsigned int x = __float_as_uint(f);
    x += 0x7fffu + ((x >> 16) & 1u);
    return (unsigned short)(x >> 16);
}

// ---------------------------------------------------------------------------
// Kernel 0: convert fp32 [R][C] -> bf16 transposed [C][R], LDS-tiled.
// Used for x (per batch: [DIN][NTOK] -> [NTOK][DIN]) and W ([DIN][DOUT] ->
// [DOUT][DIN]). Makes all downstream WMMA fragments contiguous dwords.
// ---------------------------------------------------------------------------
__global__ __launch_bounds__(256) void transpose_convert_kernel(
    const float* __restrict__ src, unsigned short* __restrict__ dst,
    int R, int C, size_t srcBatch, size_t dstBatch)
{
    __shared__ unsigned short tile[64][65];
    const float* s = src + srcBatch * blockIdx.z;
    unsigned short* d = dst + dstBatch * blockIdx.z;
    const int r0 = blockIdx.y * 64, c0 = blockIdx.x * 64;

    const int tc = threadIdx.x & 63;          // col within tile (coalesced load)
    const int tr = threadIdx.x >> 6;          // 0..3
    #pragma unroll
    for (int i = 0; i < 16; ++i) {
        int r = tr + i * 4;
        tile[r][tc] = f2bf(s[(size_t)(r0 + r) * C + (c0 + tc)]);
    }
    __syncthreads();

    const int rp = (threadIdx.x & 31) * 2;    // even row pair (contiguous in dst)
    const int cc = threadIdx.x >> 5;          // 0..7
    #pragma unroll
    for (int j = 0; j < 8; ++j) {
        int c = cc + j * 8;
        unsigned int u = (unsigned int)tile[rp][c] |
                         ((unsigned int)tile[rp + 1][c] << 16);
        *(unsigned int*)(d + (size_t)(c0 + c) * R + (r0 + rp)) = u;
    }
}

// ---------------------------------------------------------------------------
// Kernel 1: Q/K/V projections. One wave = 16x64 output strip (4 WMMA tiles
// sharing each A fragment). All operand dwords contiguous -> b128 loads.
// Q,K stored row-major [B][N][DOUT]; V stored transposed [B][DOUT][N].
// ---------------------------------------------------------------------------
__global__ __launch_bounds__(128) void qkv_gemm_kernel(
    const unsigned short* __restrict__ xbf,
    const unsigned short* __restrict__ Wtq,
    const unsigned short* __restrict__ Wtk,
    const unsigned short* __restrict__ Wtv,
    const float* __restrict__ bq, const float* __restrict__ bk,
    const float* __restrict__ bv,
    unsigned short* __restrict__ Qw,
    unsigned short* __restrict__ Kw,
    unsigned short* __restrict__ Vt)
{
    const int lane = threadIdx.x & 31;
    const int l15  = lane & 15;
    const int hi   = lane >> 4;

    const int waveId = blockIdx.x * 4 + (threadIdx.x >> 5);
    const int tilesPerBM = (NTOK / 16) * (DOUT / 64);      // 1024
    const int q  = waveId / (BQ * tilesPerBM);             // 0=Q 1=K 2=V
    int r        = waveId % (BQ * tilesPerBM);
    const int b  = r / tilesPerBM;
    r            = r % tilesPerBM;
    const int n0 = (r / (DOUT / 64)) * 16;                 // token tile
    const int c0 = (r % (DOUT / 64)) * 64;                 // out-feature strip

    const unsigned short* Wt = (q == 0) ? Wtq : (q == 1) ? Wtk : Wtv;
    const float* bias        = (q == 0) ? bq  : (q == 1) ? bk  : bv;

    const unsigned int* arow =
        (const unsigned int*)(xbf + ((size_t)b * NTOK + n0 + l15) * DIN);
    const unsigned int* brow[4];
    #pragma unroll
    for (int st = 0; st < 4; ++st)
        brow[st] = (const unsigned int*)(Wt + (size_t)(c0 + st * 16 + l15) * DIN);

    v8f zero = {};
    v8f acc[4];
    #pragma unroll
    for (int st = 0; st < 4; ++st) acc[st] = zero;

    for (int k0 = 0; k0 < DIN; k0 += 32) {
        ABFrag a;
        #pragma unroll
        for (int v = 0; v < 8; ++v) a.u[v] = arow[(k0 >> 1) + hi * 8 + v];
        #pragma unroll
        for (int st = 0; st < 4; ++st) {
            ABFrag bb;
            #pragma unroll
            for (int v = 0; v < 8; ++v) bb.u[v] = brow[st][(k0 >> 1) + hi * 8 + v];
            acc[st] = __builtin_amdgcn_wmma_f32_16x16x32_bf16(
                          false, a.v, false, bb.v, (short)0, acc[st], false, false);
        }
    }

    if (q < 2) {
        unsigned short* out = (q == 0) ? Qw : Kw;
        #pragma unroll
        for (int st = 0; st < 4; ++st) {
            const int col = c0 + st * 16 + l15;
            const float bval = bias[col];
            #pragma unroll
            for (int rI = 0; rI < 8; ++rI) {
                int row = hi * 8 + rI;
                out[((size_t)b * NTOK + (n0 + row)) * DOUT + col] =
                    f2bf(acc[st][rI] + bval);
            }
        }
    } else {
        #pragma unroll
        for (int st = 0; st < 4; ++st) {
            const int col = c0 + st * 16 + l15;
            const float bval = bias[col];
            const size_t vb = (size_t)b * DOUT * NTOK + (size_t)col * NTOK
                              + (n0 + hi * 8);
            #pragma unroll
            for (int rp = 0; rp < 4; ++rp) {
                unsigned int u =
                    (unsigned int)f2bf(acc[st][2 * rp] + bval) |
                    ((unsigned int)f2bf(acc[st][2 * rp + 1] + bval) << 16);
                *(unsigned int*)(Vt + vb + 2 * rp) = u;
            }
        }
    }
}

// ---------------------------------------------------------------------------
// Kernel 2: flash attention. One block (4 waves) per 16-token row tile.
// Scores via WMMA -> LDS fp32 -> online softmax -> P(bf16) @ V via WMMA.
// V is pre-transposed, so its B-fragments are contiguous b128 loads.
// ---------------------------------------------------------------------------
__global__ __launch_bounds__(128) void flash_attn_kernel(
    const unsigned short* __restrict__ Qw,
    const unsigned short* __restrict__ Kw,
    const unsigned short* __restrict__ Vt,
    float* __restrict__ out)
{
    __shared__ float sS[16][64];
    __shared__ float sRowM[16];
    __shared__ float sRowL[16];
    __shared__ float sAlpha[16];

    const int lane = threadIdx.x & 31;
    const int w    = threadIdx.x >> 5;     // wave 0..3
    const int l15  = lane & 15;
    const int hi   = lane >> 4;

    const int tile = blockIdx.x;                    // 0 .. B*NTOK/16-1
    const int b    = tile / (NTOK / 16);
    const int n0   = (tile % (NTOK / 16)) * 16;

    const size_t base  = (size_t)b * NTOK * DOUT;   // Q,K row-major
    const size_t vbase = (size_t)b * DOUT * NTOK;   // V transposed
    const float scale = 0.044194173824159216f;      // 1/sqrt(512)

    if (threadIdx.x < 16) { sRowM[threadIdx.x] = -3.0e38f; sRowL[threadIdx.x] = 0.0f; }
    __syncthreads();

    v8f zero = {};
    v8f oacc[8];
    #pragma unroll
    for (int t = 0; t < 8; ++t) oacc[t] = zero;

    const unsigned int* qrow =
        (const unsigned int*)(Qw + base + (size_t)(n0 + l15) * DOUT);

    for (int m0 = 0; m0 < NTOK; m0 += 64) {
        // ---- Phase A: S_w = (Q tile) @ (K rows m0+w*16..+15)^T * scale ----
        {
            const int mcol = m0 + w * 16 + l15;
            const unsigned int* krow =
                (const unsigned int*)(Kw + base + (size_t)mcol * DOUT);
            if (m0 + 64 < NTOK)
                __builtin_prefetch(krow + (64 * DOUT) / 2, 0, 0);  // next K chunk
            v8f s = zero;
            for (int d0 = 0; d0 < DOUT; d0 += 32) {
                ABFrag a, bb;
                #pragma unroll
                for (int v = 0; v < 8; ++v) a.u[v] = qrow[(d0 >> 1) + hi * 8 + v];
                #pragma unroll
                for (int v = 0; v < 8; ++v) bb.u[v] = krow[(d0 >> 1) + hi * 8 + v];
                s = __builtin_amdgcn_wmma_f32_16x16x32_bf16(
                        false, a.v, false, bb.v, (short)0, s, false, false);
            }
            #pragma unroll
            for (int rI = 0; rI < 8; ++rI) {
                int row = hi * 8 + rI;
                sS[row][w * 16 + l15] = s[rI] * scale;
            }
        }
        __syncthreads();

        // ---- Phase B: online softmax update (wave 0, one lane per row) ----
        if (threadIdx.x < 16) {
            const int row = threadIdx.x;
            float mNew = sRowM[row];
            #pragma unroll 8
            for (int c = 0; c < 64; ++c) mNew = fmaxf(mNew, sS[row][c]);
            float alpha = __expf(sRowM[row] - mNew);
            float sum = 0.0f;
            #pragma unroll 8
            for (int c = 0; c < 64; ++c) {
                float p = __expf(sS[row][c] - mNew);
                sS[row][c] = p;
                sum += p;
            }
            sRowM[row]  = mNew;
            sRowL[row]  = sRowL[row] * alpha + sum;
            sAlpha[row] = alpha;
        }
        __syncthreads();

        // ---- Phase C: rescale O, accumulate P @ V for this wave's cols ----
        #pragma unroll
        for (int t = 0; t < 8; ++t) {
            #pragma unroll
            for (int rI = 0; rI < 8; ++rI) {
                oacc[t][rI] *= sAlpha[hi * 8 + rI];
            }
        }
        #pragma unroll
        for (int kc = 0; kc < 2; ++kc) {
            ABFrag p;
            {
                const int row = l15;
                #pragma unroll
                for (int v = 0; v < 8; ++v) {
                    int k = kc * 32 + hi * 16 + 2 * v;
                    p.u[v] = (unsigned int)f2bf(sS[row][k]) |
                             ((unsigned int)f2bf(sS[row][k + 1]) << 16);
                }
            }
            const int kb = m0 + kc * 32 + hi * 16;   // key index base (B-frag K)
            #pragma unroll
            for (int t = 0; t < 8; ++t) {
                const int col = w * 128 + t * 16 + l15;
                const unsigned int* vcol =
                    (const unsigned int*)(Vt + vbase + (size_t)col * NTOK);
                ABFrag bb;
                #pragma unroll
                for (int v = 0; v < 8; ++v) bb.u[v] = vcol[(kb >> 1) + v];
                oacc[t] = __builtin_amdgcn_wmma_f32_16x16x32_bf16(
                              false, p.v, false, bb.v, (short)0, oacc[t], false, false);
            }
        }
        __syncthreads();
    }

    // ---- Finalize: divide by row sums, store fp32 ----
    #pragma unroll
    for (int rI = 0; rI < 8; ++rI) {
        int row = hi * 8 + rI;
        float linv = 1.0f / sRowL[row];
        #pragma unroll
        for (int t = 0; t < 8; ++t) {
            int col = w * 128 + t * 16 + l15;
            out[((size_t)b * NTOK + (n0 + row)) * DOUT + col] = oacc[t][rI] * linv;
        }
    }
}

extern "C" void kernel_launch(void* const* d_in, const int* in_sizes, int n_in,
                              void* d_out, int out_size, void* d_ws, size_t ws_size,
                              hipStream_t stream)
{
    const float* x  = (const float*)d_in[0];
    const float* Wq = (const float*)d_in[1];
    const float* bq = (const float*)d_in[2];
    const float* Wk = (const float*)d_in[3];
    const float* bk = (const float*)d_in[4];
    const float* Wv = (const float*)d_in[5];
    const float* bv = (const float*)d_in[6];
    float* out = (float*)d_out;

    // Workspace layout (bf16 bit-patterns as ushort):
    unsigned short* xbf = (unsigned short*)d_ws;                   // 4M elems
    unsigned short* Wtq = xbf + (size_t)BQ * NTOK * DIN;
    unsigned short* Wtk = Wtq + (size_t)DOUT * DIN;
    unsigned short* Wtv = Wtk + (size_t)DOUT * DIN;
    unsigned short* Qw  = Wtv + (size_t)DOUT * DIN;                // 4M
    unsigned short* Kw  = Qw  + (size_t)BQ * NTOK * DOUT;          // 4M
    unsigned short* Vt  = Kw  + (size_t)BQ * NTOK * DOUT;          // 4M (transposed)

    // x: [B][DIN][NTOK] -> xbf [B][NTOK][DIN]
    transpose_convert_kernel<<<dim3(NTOK / 64, DIN / 64, BQ), 256, 0, stream>>>(
        x, xbf, DIN, NTOK, (size_t)DIN * NTOK, (size_t)NTOK * DIN);
    // W: [DIN][DOUT] -> Wt [DOUT][DIN]
    transpose_convert_kernel<<<dim3(DOUT / 64, DIN / 64, 1), 256, 0, stream>>>(
        Wq, Wtq, DIN, DOUT, 0, 0);
    transpose_convert_kernel<<<dim3(DOUT / 64, DIN / 64, 1), 256, 0, stream>>>(
        Wk, Wtk, DIN, DOUT, 0, 0);
    transpose_convert_kernel<<<dim3(DOUT / 64, DIN / 64, 1), 256, 0, stream>>>(
        Wv, Wtv, DIN, DOUT, 0, 0);

    const int totalWaves = 3 * BQ * (NTOK / 16) * (DOUT / 64);     // 12288
    qkv_gemm_kernel<<<totalWaves / 4, 128, 0, stream>>>(
        xbf, Wtq, Wtk, Wtv, bq, bk, bv, Qw, Kw, Vt);

    flash_attn_kernel<<<BQ * (NTOK / 16), 128, 0, stream>>>(Qw, Kw, Vt, out);
}
